// SCMRSSM_52682068853313
// MI455X (gfx1250) — matrix operand order repaired and checked by reading, not compiled, not run on hardware
//
#include <hip/hip_runtime.h>
#include <hip/hip_bf16.h>

// ---------------------------------------------------------------------------
// DreamerV3 RSSM observe() for MI455X (gfx1250, wave32, WMMA).
// Weights converted once to bf16 into d_ws (156.5 MB -> fits 192 MB L2),
// all 6 per-step GEMMs run v_wmma_f32_16x16x32_bf16 with fp32 accumulation.
// A-matrix chunks staged to LDS via GLOBAL_LOAD_ASYNC_TO_LDS_B128 (ASYNCcnt
// path); plain ld/st fallback if the builtin is absent.
// ---------------------------------------------------------------------------

typedef __attribute__((ext_vector_type(16))) __bf16          v16bf;
typedef __attribute__((ext_vector_type(8)))  float           v8f;
typedef __attribute__((ext_vector_type(8)))  unsigned short  ushort8;
typedef __attribute__((ext_vector_type(16))) unsigned short  ushort16;
typedef __attribute__((ext_vector_type(4)))  int             v4i;

#define Bsz    32
#define Tlen   64
#define STOCH  1024
#define DETER  4096
#define HIDDEN 1024
#define ACTD   128
#define EMBD   1536
#define OUTW   (DETER + 6 * STOCH)   // 10240
#define KCHUNK 128
#define GEMM_THREADS 128             // 4 waves = one WGP's 4 SIMD32s

#if __has_builtin(__builtin_amdgcn_global_load_async_to_lds_b128)
#define USE_ASYNC_LDS 1
typedef __attribute__((address_space(1))) v4i* g1_v4i_p;   // global int4*
typedef __attribute__((address_space(3))) v4i* l3_v4i_p;   // LDS int4*
#endif

__device__ __forceinline__ void wait_async0() {
#if __has_builtin(__builtin_amdgcn_s_wait_asynccnt)
  __builtin_amdgcn_s_wait_asynccnt(0);
#else
  asm volatile("s_wait_asynccnt 0x0" ::: "memory");
#endif
}

__device__ __forceinline__ unsigned short f2bf(float f) {
  unsigned int u = __builtin_bit_cast(unsigned int, f);
  unsigned int r = u + 0x7FFFu + ((u >> 16) & 1u);   // round-to-nearest-even
  return (unsigned short)(r >> 16);
}
__device__ __forceinline__ float sigmoidf_(float x) { return 1.0f / (1.0f + __expf(-x)); }
__device__ __forceinline__ float softplusf_(float x) {
  return (x > 20.0f) ? x : log1pf(__expf(x));
}

// ---------------------------------------------------------------------------
// Generic skinny GEMM:  C[32,N] = A[32,K] @ W[K,N]   (A,W bf16, C fp32)
// 128 threads = 4 waves; each wave owns one 16-wide column strip and both
// 16-row halves (two accumulators share each B fragment).
// A staged through LDS per 128-wide K chunk; W streams from L2 with a
// running per-lane row pointer (no per-iteration 64-bit multiplies).
// ---------------------------------------------------------------------------
__global__ __launch_bounds__(GEMM_THREADS) void gemm_a32_bf16(
    const unsigned short* __restrict__ A,   // [32, lda] bf16 bits
    const unsigned short* __restrict__ W,   // [K, N]    bf16 bits
    float* __restrict__ C,                  // [32, N]   fp32
    int lda, int K, int N)
{
  __shared__ unsigned short lA[32 * KCHUNK];

  const int wave = threadIdx.x >> 5;
  const int lane = threadIdx.x & 31;
  const int ncol = (blockIdx.x * 4 + wave) * 16;

  v8f acc0 = {};
  v8f acc1 = {};

  const int r0 = lane & 15;          // row within 16-row tile
  const int kb = (lane >> 4) * 8;    // ISA 16-bit A layout: hi lanes start at K=8

  // running per-lane weight pointer: column ncol + (lane & 15), row = k0 + kk
  const unsigned short* __restrict__ Wrow = W + ncol + (lane & 15);
  const size_t wStep = (size_t)32 * N;

  for (int k0 = 0; k0 < K; k0 += KCHUNK) {
    // ---- stage A[32, k0:k0+128] into LDS (row-major [32][KCHUNK]) ----
#if defined(USE_ASYNC_LDS) && defined(__HIP_DEVICE_COMPILE__)
    for (int idx = threadIdx.x; idx < 32 * KCHUNK / 8; idx += GEMM_THREADS) {
      int r = idx / (KCHUNK / 8);
      int c = (idx % (KCHUNK / 8)) * 8;
      __builtin_amdgcn_global_load_async_to_lds_b128(
          (g1_v4i_p)&A[(size_t)r * lda + k0 + c],
          (l3_v4i_p)&lA[r * KCHUNK + c], 0, 0);
    }
    wait_async0();
#else
    for (int idx = threadIdx.x; idx < 32 * KCHUNK / 8; idx += GEMM_THREADS) {
      int r = idx / (KCHUNK / 8);
      int c = (idx % (KCHUNK / 8)) * 8;
      *(ushort8*)&lA[r * KCHUNK + c] =
          *(const ushort8*)&A[(size_t)r * lda + k0 + c];
    }
#endif
    __syncthreads();

    if (k0 + KCHUNK < K)  // pull next weight chunk toward L2/L0
      __builtin_prefetch(Wrow + (size_t)KCHUNK * N, 0, 0);

    for (int kk = 0; kk < KCHUNK; kk += 32) {
      // B fragment: lane = N column, K {kb..kb+7, kb+16..kb+23} (same order as A)
      ushort16 bw;
#pragma unroll
      for (int s = 0; s < 16; ++s)
        bw[s] = Wrow[(size_t)(kb + (s & 7) + (s >> 3) * 16) * N];
      Wrow += wStep;
      v16bf bf = __builtin_bit_cast(v16bf, bw);

      // A fragments per ISA layout: lanes<16 hold K {0..7,16..23}, hi lanes {8..15,24..31}
      ushort8 a0lo = *(const ushort8*)&lA[r0 * KCHUNK + kk + kb];
      ushort8 a0hi = *(const ushort8*)&lA[r0 * KCHUNK + kk + kb + 16];
      ushort8 a1lo = *(const ushort8*)&lA[(16 + r0) * KCHUNK + kk + kb];
      ushort8 a1hi = *(const ushort8*)&lA[(16 + r0) * KCHUNK + kk + kb + 16];
      ushort16 u0 = __builtin_shufflevector(a0lo, a0hi, 0,1,2,3,4,5,6,7,8,9,10,11,12,13,14,15);
      ushort16 u1 = __builtin_shufflevector(a1lo, a1hi, 0,1,2,3,4,5,6,7,8,9,10,11,12,13,14,15);
      v16bf af0 = __builtin_bit_cast(v16bf, u0);
      v16bf af1 = __builtin_bit_cast(v16bf, u1);

      acc0 = __builtin_amdgcn_wmma_f32_16x16x32_bf16(false, af0, false, bf,
                                                     (short)0, acc0, false, false);
      acc1 = __builtin_amdgcn_wmma_f32_16x16x32_bf16(false, af1, false, bf,
                                                     (short)0, acc1, false, false);
    }
    __syncthreads();
  }

  // C/D layout: VGPR i -> row i (+8 for hi lanes), col = lane & 15
  const int col = ncol + (lane & 15);
  const int rb  = (lane < 16) ? 0 : 8;
#pragma unroll
  for (int i = 0; i < 8; ++i) {
    C[(size_t)(rb + i) * N + col]      = acc0[i];
    C[(size_t)(16 + rb + i) * N + col] = acc1[i];
  }
}

// ---------------------------------------------------------------------------
// Block-wide two-value reduction (sum, sum of squares)
// ---------------------------------------------------------------------------
__device__ __forceinline__ void block_reduce2(float& s, float& s2) {
  __shared__ float sh[256], sh2[256];
  sh[threadIdx.x] = s; sh2[threadIdx.x] = s2;
  __syncthreads();
  for (int o = 128; o > 0; o >>= 1) {
    if (threadIdx.x < o) { sh[threadIdx.x] += sh[threadIdx.x + o];
                           sh2[threadIdx.x] += sh2[threadIdx.x + o]; }
    __syncthreads();
  }
  s = sh[0]; s2 = sh2[0];
}

// LN(1024) + SiLU -> bf16, one block per batch row
__global__ __launch_bounds__(256) void ln_silu_1024(
    const float* __restrict__ raw, const float* __restrict__ g,
    const float* __restrict__ be, unsigned short* __restrict__ dst, int dstStride)
{
  const int b = blockIdx.x;
  const float* r = raw + (size_t)b * 1024;
  float s = 0.f, s2 = 0.f;
  for (int i = threadIdx.x; i < 1024; i += 256) { float v = r[i]; s += v; s2 += v * v; }
  block_reduce2(s, s2);
  const float mean = s * (1.0f / 1024.0f);
  const float var  = s2 * (1.0f / 1024.0f) - mean * mean;
  const float rs   = rsqrtf(var + 1e-3f);
  for (int i = threadIdx.x; i < 1024; i += 256) {
    float x = (r[i] - mean) * rs * g[i] + be[i];
    float y = x * sigmoidf_(x);                       // SiLU
    dst[(size_t)b * dstStride + i] = f2bf(y);
  }
}

// GRU: LN(12288) + gates + deter update; also stages deter (bf16) into the
// GRU-A and obs-A concat buffers, writes deter to output, and stages embed_t.
__global__ __launch_bounds__(256) void gru_update(
    const float* __restrict__ raw,           // [32,12288]
    const float* __restrict__ g, const float* __restrict__ be,
    float* __restrict__ deterF,              // [32,4096] state
    unsigned short* __restrict__ gruA,       // [32,5120]  (cols 1024.. get deter)
    unsigned short* __restrict__ obsA,       // [32,5632]  (deter | embed)
    const float* __restrict__ embed,         // [32,64,1536]
    float* __restrict__ out, int t)
{
  const int b = blockIdx.x;
  const float* r = raw + (size_t)b * 12288;
  float s = 0.f, s2 = 0.f;
  for (int i = threadIdx.x; i < 12288; i += 256) { float v = r[i]; s += v; s2 += v * v; }
  block_reduce2(s, s2);
  const float mean = s * (1.0f / 12288.0f);
  const float var  = s2 * (1.0f / 12288.0f) - mean * mean;
  const float rs   = rsqrtf(var + 1e-3f);

  const size_t ob = ((size_t)b * Tlen + t) * OUTW;
  for (int j = threadIdx.x; j < DETER; j += 256) {
    float rr = (r[j]           - mean) * rs * g[j]           + be[j];
    float cc = (r[4096 + j]    - mean) * rs * g[4096 + j]    + be[4096 + j];
    float uu = (r[8192 + j]    - mean) * rs * g[8192 + j]    + be[8192 + j];
    float reset = sigmoidf_(rr);
    float cand  = tanhf(reset * cc);
    float upd   = sigmoidf_(uu - 1.0f);
    float dn    = upd * cand + (1.0f - upd) * deterF[(size_t)b * DETER + j];
    deterF[(size_t)b * DETER + j] = dn;
    unsigned short h = f2bf(dn);
    gruA[(size_t)b * 5120 + 1024 + j] = h;
    obsA[(size_t)b * 5632 + j]        = h;
    out[ob + j] = dn;
  }
  const float* e = embed + ((size_t)b * Tlen + t) * EMBD;
  for (int j = threadIdx.x; j < EMBD; j += 256)
    obsA[(size_t)b * 5632 + DETER + j] = f2bf(e[j]);
}

// Suff-stats: bias, softplus(+0.1), sample; pack outputs; build next-step A1.
__global__ __launch_bounds__(256) void finalize_step(
    const float* __restrict__ spRaw, const float* __restrict__ sqRaw, // [32,2048]
    const float* __restrict__ b_ims, const float* __restrict__ b_obs,
    const float* __restrict__ noiseP, const float* __restrict__ noiseQ,
    const float* __restrict__ action,
    float* __restrict__ out, unsigned short* __restrict__ A1, int t)
{
  const int b = blockIdx.x;
  const size_t ob = ((size_t)b * Tlen + t) * OUTW;
  for (int i = threadIdx.x; i < STOCH; i += 256) {
    float mp = spRaw[(size_t)b * 2048 + i] + b_ims[i];
    float sp = softplusf_(spRaw[(size_t)b * 2048 + 1024 + i] + b_ims[1024 + i]) + 0.1f;
    float zp = mp + sp * noiseP[((size_t)b * Tlen + t) * STOCH + i];
    float mq = sqRaw[(size_t)b * 2048 + i] + b_obs[i];
    float sq = softplusf_(sqRaw[(size_t)b * 2048 + 1024 + i] + b_obs[1024 + i]) + 0.1f;
    float zq = mq + sq * noiseQ[((size_t)b * Tlen + t) * STOCH + i];
    out[ob + 4096 + i] = zq;   // stoch_q
    out[ob + 5120 + i] = mq;   // mean_q
    out[ob + 6144 + i] = sq;   // std_q
    out[ob + 7168 + i] = zp;   // stoch_p
    out[ob + 8192 + i] = mp;   // mean_p
    out[ob + 9216 + i] = sp;   // std_p
    A1[(size_t)b * 1152 + i] = f2bf(zq);             // next-step stoch input
    if (t + 1 < Tlen && i < ACTD)
      A1[(size_t)b * 1152 + 1024 + i] =
          f2bf(action[((size_t)b * Tlen + t + 1) * ACTD + i]);
  }
}

__global__ __launch_bounds__(256) void cvt_f32_bf16(
    const float* __restrict__ src, unsigned short* __restrict__ dst, size_t n)
{
  size_t i = (size_t)blockIdx.x * 256 + threadIdx.x;
  size_t stride = (size_t)gridDim.x * 256;
  for (; i < n; i += stride) dst[i] = f2bf(src[i]);
}

__global__ __launch_bounds__(256) void init_state(
    const float* __restrict__ init_stoch, const float* __restrict__ init_deter,
    const float* __restrict__ action,
    unsigned short* __restrict__ A1, unsigned short* __restrict__ gruA,
    float* __restrict__ deterF)
{
  const int b = blockIdx.x;
  for (int i = threadIdx.x; i < STOCH; i += 256)
    A1[(size_t)b * 1152 + i] = f2bf(init_stoch[(size_t)b * STOCH + i]);
  for (int i = threadIdx.x; i < ACTD; i += 256)
    A1[(size_t)b * 1152 + 1024 + i] = f2bf(action[((size_t)b * Tlen + 0) * ACTD + i]);
  for (int j = threadIdx.x; j < DETER; j += 256) {
    float d = init_deter[(size_t)b * DETER + j];
    deterF[(size_t)b * DETER + j] = d;
    gruA[(size_t)b * 5120 + 1024 + j] = f2bf(d);
  }
}

// ---------------------------------------------------------------------------
extern "C" void kernel_launch(void* const* d_in, const int* in_sizes, int n_in,
                              void* d_out, int out_size, void* d_ws, size_t ws_size,
                              hipStream_t stream) {
  (void)in_sizes; (void)n_in; (void)out_size; (void)ws_size;
  const float* embed      = (const float*)d_in[0];
  const float* action     = (const float*)d_in[1];
  /* d_in[2] = is_first, all false -> unused */
  const float* init_deter = (const float*)d_in[3];
  const float* init_stoch = (const float*)d_in[4];
  const float* noiseP     = (const float*)d_in[5];
  const float* noiseQ     = (const float*)d_in[6];
  const float* W_in   = (const float*)d_in[7];
  const float* g_in   = (const float*)d_in[8];
  const float* b_in   = (const float*)d_in[9];
  const float* W_gru  = (const float*)d_in[10];
  const float* g_gru  = (const float*)d_in[11];
  const float* b_gru  = (const float*)d_in[12];
  const float* W_io   = (const float*)d_in[13];
  const float* g_io   = (const float*)d_in[14];
  const float* b_io   = (const float*)d_in[15];
  const float* W_oo   = (const float*)d_in[16];
  const float* g_oo   = (const float*)d_in[17];
  const float* b_oo   = (const float*)d_in[18];
  const float* W_ims  = (const float*)d_in[19];
  const float* b_ims  = (const float*)d_in[20];
  const float* W_obs  = (const float*)d_in[21];
  const float* b_obs  = (const float*)d_in[22];
  float* out = (float*)d_out;

  // -------- workspace carve-up (256B aligned) --------
  char* p = (char*)d_ws;
  auto alloc = [&](size_t bytes) -> char* {
    char* r = p; p += (bytes + 255) & ~(size_t)255; return r;
  };
  unsigned short* wInB  = (unsigned short*)alloc((size_t)1152 * 1024 * 2);
  unsigned short* wGruB = (unsigned short*)alloc((size_t)5120 * 12288 * 2);
  unsigned short* wIoB  = (unsigned short*)alloc((size_t)4096 * 1024 * 2);
  unsigned short* wOoB  = (unsigned short*)alloc((size_t)5632 * 1024 * 2);
  unsigned short* wImsB = (unsigned short*)alloc((size_t)1024 * 2048 * 2);
  unsigned short* wObsB = (unsigned short*)alloc((size_t)1024 * 2048 * 2);
  unsigned short* A1    = (unsigned short*)alloc((size_t)32 * 1152 * 2);  // [stoch|action]
  unsigned short* gruA  = (unsigned short*)alloc((size_t)32 * 5120 * 2);  // [x|deter]
  unsigned short* obsA  = (unsigned short*)alloc((size_t)32 * 5632 * 2);  // [deter|embed]
  unsigned short* hpB   = (unsigned short*)alloc((size_t)32 * 1024 * 2);
  unsigned short* hqB   = (unsigned short*)alloc((size_t)32 * 1024 * 2);
  float* imgRaw = (float*)alloc((size_t)32 * 1024 * 4);
  float* gruRaw = (float*)alloc((size_t)32 * 12288 * 4);
  float* hpRaw  = (float*)alloc((size_t)32 * 1024 * 4);
  float* hqRaw  = (float*)alloc((size_t)32 * 1024 * 4);
  float* spRaw  = (float*)alloc((size_t)32 * 2048 * 4);
  float* sqRaw  = (float*)alloc((size_t)32 * 2048 * 4);
  float* deterF = (float*)alloc((size_t)32 * 4096 * 4);

  // -------- prologue: bf16 weight residency (fits 192MB L2) --------
  cvt_f32_bf16<<<1024, 256, 0, stream>>>(W_in,  wInB,  (size_t)1152 * 1024);
  cvt_f32_bf16<<<4096, 256, 0, stream>>>(W_gru, wGruB, (size_t)5120 * 12288);
  cvt_f32_bf16<<<1024, 256, 0, stream>>>(W_io,  wIoB,  (size_t)4096 * 1024);
  cvt_f32_bf16<<<1024, 256, 0, stream>>>(W_oo,  wOoB,  (size_t)5632 * 1024);
  cvt_f32_bf16<<<1024, 256, 0, stream>>>(W_ims, wImsB, (size_t)1024 * 2048);
  cvt_f32_bf16<<<1024, 256, 0, stream>>>(W_obs, wObsB, (size_t)1024 * 2048);
  init_state<<<32, 256, 0, stream>>>(init_stoch, init_deter, action, A1, gruA, deterF);

  // -------- sequential scan over T --------
  for (int t = 0; t < Tlen; ++t) {
    // img_in: x = SiLU(LN(A1 @ W_in))
    gemm_a32_bf16<<<1024 / 64, GEMM_THREADS, 0, stream>>>(A1, wInB, imgRaw, 1152, 1152, 1024);
    ln_silu_1024<<<32, 256, 0, stream>>>(imgRaw, g_in, b_in, gruA, 5120);
    // GRU: parts = LN([x|deter] @ W_gru) ; gates ; deter update
    gemm_a32_bf16<<<12288 / 64, GEMM_THREADS, 0, stream>>>(gruA, wGruB, gruRaw, 5120, 5120, 12288);
    gru_update<<<32, 256, 0, stream>>>(gruRaw, g_gru, b_gru, deterF, gruA, obsA,
                                       embed, out, t);
    // img_out / obs_out heads
    gemm_a32_bf16<<<1024 / 64, GEMM_THREADS, 0, stream>>>(obsA, wIoB, hpRaw, 5632, 4096, 1024);
    gemm_a32_bf16<<<1024 / 64, GEMM_THREADS, 0, stream>>>(obsA, wOoB, hqRaw, 5632, 5632, 1024);
    ln_silu_1024<<<32, 256, 0, stream>>>(hpRaw, g_io, b_io, hpB, 1024);
    ln_silu_1024<<<32, 256, 0, stream>>>(hqRaw, g_oo, b_oo, hqB, 1024);
    // sufficient statistics
    gemm_a32_bf16<<<2048 / 64, GEMM_THREADS, 0, stream>>>(hpB, wImsB, spRaw, 1024, 1024, 2048);
    gemm_a32_bf16<<<2048 / 64, GEMM_THREADS, 0, stream>>>(hqB, wObsB, sqRaw, 1024, 1024, 2048);
    finalize_step<<<32, 256, 0, stream>>>(spRaw, sqRaw, b_ims, b_obs,
                                          noiseP, noiseQ, action, out, A1, t);
  }
}
